// RNN_21732534518291
// MI455X (gfx1250) — matrix operand order repaired and checked by reading, not compiled, not run on hardware
//
#include <hip/hip_runtime.h>

typedef __attribute__((ext_vector_type(16))) _Float16 v16h;
typedef __attribute__((ext_vector_type(2)))  __fp16   fp16x2;
typedef __attribute__((ext_vector_type(8)))  float    v8f;
typedef __attribute__((ext_vector_type(2)))  unsigned u2;

#define T_LEN  1024
#define NBATCH 512
#define HID    64

#if __has_builtin(__builtin_amdgcn_tanhf)
// CDNA5 v_tanh_f32 transcendental — single trans op on the recurrence path.
__device__ __forceinline__ float fast_tanh(float v) {
    return __builtin_amdgcn_tanhf(v);
}
#else
__device__ __forceinline__ float fast_tanh(float v) {
    float e = __expf(2.0f * v);
    float r = __builtin_amdgcn_rcpf(e + 1.0f);
    return 1.0f - 2.0f * r;
}
#endif

// One v_cvt_pk_rtz_f16_f32 per pair.
__device__ __forceinline__ unsigned pack2h(float a, float b) {
    union { unsigned u; fp16x2 p; } z;
    z.p = __builtin_amdgcn_cvt_pkrtz(a, b);
    return z.u;
}

// Half-wave fragment swap (v_permlane16_swap_b32):
//   new a: lanes 0-15 = own a,            lanes 16-31 = b of (lane-16)
//   new b: lanes 0-15 = a of (lane+16),   lanes 16-31 = own b
#if __has_builtin(__builtin_amdgcn_permlane16_swap)
#define HAVE_PL16SWAP 1
__device__ __forceinline__ void half_swap(unsigned& a, unsigned& b) {
    u2 r = __builtin_amdgcn_permlane16_swap(a, b, false, false);
    a = r[0];
    b = r[1];
}
// Value held by the hi-partner lane (lane ^ 16), without touching LDS.
__device__ __forceinline__ float partner_f32(float v, int hi) {
    union { unsigned u; float f; } z;
    z.f = v;
    u2 r = __builtin_amdgcn_permlane16_swap(z.u, z.u, false, false);
    z.u = hi ? r[0] : r[1];  // lanes>=16: r[0] = partner; lanes<16: r[1]
    return z.f;
}
#else
__device__ __forceinline__ void half_swap(unsigned& a, unsigned& b) {
    const int hi = (threadIdx.x & 31) >> 4;
    unsigned sa = (unsigned)__shfl_xor((int)a, 16, 32);
    unsigned sb = (unsigned)__shfl_xor((int)b, 16, 32);
    a = hi ? sb : a;
    b = hi ? b : sa;
}
__device__ __forceinline__ float partner_f32(float v, int hi) {
    (void)hi;
    return __shfl_xor(v, 16, 32);
}
#endif

union BFrag { v16h v; unsigned d[8]; };

// Transposed recurrence: Ht(64x16, hidden x batch) per wave.
//   Ht_new = tanh(xp^T + W_hh @ Ht)
// A = W_hh tiles (loop-invariant VGPRs). B = Ht (2 fragments of 32x16).
// Bias rides in the WMMA C operand. D->B relayout and the output reduction
// are pure register traffic (v_permlane16_swap_b32): zero LDS ops in the
// loop. x[t+1] is prefetched one iteration ahead so its latency hides
// behind the WMMA + tanh work of step t.
__global__ __launch_bounds__(32) void rnn_fused_kernel(
    const float* __restrict__ x,      // (T, B, 1)
    const float* __restrict__ W_ih,   // (H, 1)
    const float* __restrict__ W_hh,   // (H, H) row-major
    const float* __restrict__ b_ih,   // (H)
    const float* __restrict__ b_hh,   // (H)
    const float* __restrict__ W_out,  // (1, H)
    const float* __restrict__ b_out,  // (1)
    float* __restrict__ out)          // (T, B, 1)
{
    const int lane  = threadIdx.x & 31;
    const int ln    = lane & 15;        // batch column within the 16-wide tile
    const int hi    = lane >> 4;        // half-wave select
    const int bbase = blockIdx.x * 16;  // batch-tile base

    // ---- A fragments of W_hh: A[m][kc] = rows 16m..16m+15, cols 32kc..+31.
    // A element i of lane: row = 16m+ln, k = 32kc + 16*hi + i.
    v16h A[4][2];
    #pragma unroll
    for (int m = 0; m < 4; ++m)
      #pragma unroll
      for (int kc = 0; kc < 2; ++kc) {
        const float* wrow = &W_hh[(16 * m + ln) * HID + 32 * kc + 16 * hi];
        #pragma unroll
        for (int i = 0; i < 16; ++i) {
          const int k = i;
          A[m][kc][i] = (_Float16)wrow[k];
        }
      }

    // Per-lane hidden-row constants; element (m,r) is hidden index
    // h = 16m + 8*hi + r (the D-fragment row mapping).
    v8f   biasv[4];                 // rides in WMMA C operand
    float wihc[4][8], woutc[4][8];
    #pragma unroll
    for (int m = 0; m < 4; ++m)
      #pragma unroll
      for (int r = 0; r < 8; ++r) {
        const int h = 16 * m + 8 * hi + r;
        biasv[m][r] = b_ih[h] + b_hh[h];
        wihc[m][r]  = W_ih[h];
        woutc[m][r] = W_out[h];
      }
    const float bout = b_out[0];

    // h0 = 0.
    BFrag B0, B1;
    #pragma unroll
    for (int j = 0; j < 8; ++j) { B0.d[j] = 0u; B1.d[j] = 0u; }

    // Prime the x pipeline: xt holds x[t] for the current iteration.
    const int xbase = bbase + ln;
    float xt = x[xbase];  // t = 0

    for (int t = 0; t < T_LEN; ++t) {
      // Prefetch x for the next step (wrap at the end: branchless, in-bounds).
      const int tn = (t + 1) & (T_LEN - 1);
      const float xt_next = x[tn * NBATCH + xbase];

      unsigned Dm[4][4];   // packed f16 pairs: Dm[m][j] = h_new(k=16m+8hi+2j, +1)
      float pse[4], pso[4];  // per-m even/odd output-projection partials

      #pragma unroll
      for (int m = 0; m < 4; ++m) {
        v8f acc;
        acc = __builtin_amdgcn_wmma_f32_16x16x32_f16(
            false, A[m][0], false, B0.v, (short)0, biasv[m], false, false);
        acc = __builtin_amdgcn_wmma_f32_16x16x32_f16(
            false, A[m][1], false, B1.v, (short)0, acc, false, false);

        float th[8];
        float pe = 0.f, po = 0.f;  // two chains -> v_dual_fmac
        #pragma unroll
        for (int r = 0; r < 8; ++r) {
          const float hv = acc[r] + xt * wihc[m][r];  // bias already in acc
          th[r] = fast_tanh(hv);
          if (r & 1) po += th[r] * woutc[m][r];
          else       pe += th[r] * woutc[m][r];
        }
        pse[m] = pe;
        pso[m] = po;
        #pragma unroll
        for (int j = 0; j < 4; ++j)
          Dm[m][j] = pack2h(th[2 * j], th[2 * j + 1]);
      }

      // ---- D -> B relayout via half-wave swaps (register-only).
      // B frag0 covers k = 16*hi + 0..15; frag1 covers k = 32 + 16*hi + 0..15.
      #pragma unroll
      for (int j = 0; j < 4; ++j) {
        unsigned a0 = Dm[0][j], b0 = Dm[1][j];
        half_swap(a0, b0);
        B0.d[j] = a0;        // hi=0: k 0..7   | hi=1: k 16..23
        B0.d[4 + j] = b0;    // hi=0: k 8..15  | hi=1: k 24..31
        unsigned a1 = Dm[2][j], b1 = Dm[3][j];
        half_swap(a1, b1);
        B1.d[j] = a1;        // hi=0: k 32..39 | hi=1: k 48..55
        B1.d[4 + j] = b1;    // hi=0: k 40..47 | hi=1: k 56..63
      }

      // ---- out[t, b]: this lane + its hi-partner cover all 64 hidden rows.
      float psum = ((pse[0] + pso[0]) + (pse[1] + pso[1])) +
                   ((pse[2] + pso[2]) + (pse[3] + pso[3]));
      psum += partner_f32(psum, hi);
      if (hi == 0)
        out[t * NBATCH + bbase + ln] = psum + bout;  // 16 coalesced b32 stores

      xt = xt_next;
    }
}

extern "C" void kernel_launch(void* const* d_in, const int* in_sizes, int n_in,
                              void* d_out, int out_size, void* d_ws, size_t ws_size,
                              hipStream_t stream) {
    (void)in_sizes; (void)n_in; (void)d_ws; (void)ws_size; (void)out_size;
    const float* x     = (const float*)d_in[0];
    const float* W_ih  = (const float*)d_in[1];
    const float* W_hh  = (const float*)d_in[2];
    const float* b_ih  = (const float*)d_in[3];
    const float* b_hh  = (const float*)d_in[4];
    const float* W_out = (const float*)d_in[5];
    const float* b_out = (const float*)d_in[6];
    float* out = (float*)d_out;

    dim3 grid(NBATCH / 16);  // 32 batch tiles, one wave each
    dim3 block(32);
    rnn_fused_kernel<<<grid, block, 0, stream>>>(x, W_ih, W_hh, b_ih, b_hh,
                                                 W_out, b_out, out);
}
